// CausalMQA_53687091200234
// MI455X (gfx1250) — matrix operand order repaired and checked by reading, not compiled, not run on hardware
//
#include <hip/hip_runtime.h>

#define B_  2
#define T_  2048
#define C_  2048
#define NH_ 16
#define HD_ 128

typedef __attribute__((ext_vector_type(8)))  __bf16 bf16x8;
typedef __attribute__((ext_vector_type(16))) __bf16 bf16x16;
typedef __attribute__((ext_vector_type(8)))  float  f32x8;

// gcc-style vector to match the async-to-LDS builtin's parameter type exactly
typedef int i32x4 __attribute__((vector_size(16)));
typedef __attribute__((address_space(1))) i32x4 gas_i32x4;
typedef __attribute__((address_space(3))) i32x4 las_i32x4;

union ABf { bf16x16 v16; bf16x8 v8[2]; };

static __device__ inline f32x8 zero8() {
  f32x8 z = {0.f,0.f,0.f,0.f,0.f,0.f,0.f,0.f};
  return z;
}

#if __has_builtin(__builtin_amdgcn_global_load_async_to_lds_b128)
#define HAVE_ASYNC_LDS 1
#else
#define HAVE_ASYNC_LDS 0
#endif

static __device__ inline void wait_async0() {
#if __has_builtin(__builtin_amdgcn_s_wait_asynccnt)
  __builtin_amdgcn_s_wait_asynccnt(0);
#else
  asm volatile("s_wait_asynccnt 0x0" ::: "memory");
#endif
}

// hard scheduling fence: nothing moves across (shapes the software pipeline)
static __device__ inline void sched_fence() {
#if __has_builtin(__builtin_amdgcn_sched_barrier)
  __builtin_amdgcn_sched_barrier(0);
#endif
}

// ---------------------------------------------------------------------------
// fp32 -> bf16 elementwise conversion (grid-stride)
// ---------------------------------------------------------------------------
__global__ void cvt_f32_bf16(const float* __restrict__ in, __bf16* __restrict__ out, int n) {
  int i = blockIdx.x * blockDim.x + threadIdx.x;
  const int stride = gridDim.x * blockDim.x;
  for (; i < n; i += stride) out[i] = (__bf16)in[i];
}

// ---------------------------------------------------------------------------
// Y[M][N] = scale * (A[M][K] @ W[N][K]^T), bf16 in, fp32 accum, bf16 out.
// Block = 256 threads = 8 waves; block tile 256(M) x 64(N); wave tile 32 x 64.
// K-step 32 via v_wmma_f32_16x16x32_bf16; ping-pong register buffers unrolled
// by 2, with sched_barrier(0) fences pinning the load/mma phase order so the
// machine scheduler cannot re-sink loads next to their consuming WMMAs.
// ---------------------------------------------------------------------------
__global__ __launch_bounds__(256) void gemm_bf16_nt(
    const __bf16* __restrict__ A, const __bf16* __restrict__ W,
    __bf16* __restrict__ Y, int N, int K, float scale)
{
  const int lane = threadIdx.x & 31;
  const int wv   = threadIdx.x >> 5;
  const int hf   = lane >> 4;       // which 16-lane half
  const int l16  = lane & 15;
  const int m0   = blockIdx.y * 256 + wv * 32;
  const int n0   = blockIdx.x * 64;

  f32x8 acc[2][4];
  #pragma unroll
  for (int mt = 0; mt < 2; ++mt)
    #pragma unroll
    for (int nt = 0; nt < 4; ++nt) acc[mt][nt] = zero8();

  const __bf16* arow0 = A + (size_t)(m0 + l16) * K;        // A-frag rows m0..+15
  const __bf16* arow1 = A + (size_t)(m0 + 16 + l16) * K;   // A-frag rows m0+16..+31
  const __bf16* wbase = W + (size_t)(n0 + l16) * K;        // B-frag: lane = col

  ABf a0[2], a1[2];
  bf16x16 bb[2][4];

  auto loadFrags = [&](int buf, int kk) {
    a0[buf].v8[0] = *(const bf16x8*)(arow0 + kk + hf * 16);
    a0[buf].v8[1] = *(const bf16x8*)(arow0 + kk + hf * 16 + 8);
    a1[buf].v8[0] = *(const bf16x8*)(arow1 + kk + hf * 16);
    a1[buf].v8[1] = *(const bf16x8*)(arow1 + kk + hf * 16 + 8);
    #pragma unroll
    for (int nt = 0; nt < 4; ++nt)
      bb[buf][nt] = *(const bf16x16*)(wbase + (size_t)nt * 16 * K + kk + hf * 16);
  };
  auto mmaFrags = [&](int buf) {
    #pragma unroll
    for (int nt = 0; nt < 4; ++nt) {
      acc[0][nt] = __builtin_amdgcn_wmma_f32_16x16x32_bf16(
          false, a0[buf].v16, false, bb[buf][nt], (short)0, acc[0][nt], false, false);
      acc[1][nt] = __builtin_amdgcn_wmma_f32_16x16x32_bf16(
          false, a1[buf].v16, false, bb[buf][nt], (short)0, acc[1][nt], false, false);
    }
  };

  loadFrags(0, 0);                       // prologue
  for (int k0 = 0; k0 < K; k0 += 64) {   // K multiple of 64
    __builtin_prefetch(arow0 + k0 + 512, 0, 1);
    const int k1 = (k0 + 32 < K) ? k0 + 32 : k0;   // clamped: harmless reload
    loadFrags(1, k1);
    sched_fence();
    mmaFrags(0);
    sched_fence();
    const int k2 = (k0 + 64 < K) ? k0 + 64 : k0;
    loadFrags(0, k2);
    sched_fence();
    mmaFrags(1);
    sched_fence();
  }

  // C layout: row = hf*8 + j (vgpr j), col = l16
  #pragma unroll
  for (int mt = 0; mt < 2; ++mt)
    #pragma unroll
    for (int nt = 0; nt < 4; ++nt)
      #pragma unroll
      for (int j = 0; j < 8; ++j) {
        const int row = m0 + mt * 16 + hf * 8 + j;
        Y[(size_t)row * N + n0 + nt * 16 + l16] = (__bf16)(acc[mt][nt][j] * scale);
      }
}

// ---------------------------------------------------------------------------
// Fused causal flash attention, double-buffered LDS staging.
// q:  [B*T, 2048] bf16 (pre-scaled by 1/sqrt(hd)); head h = cols h*128..+127
// kv: [B*T, 256]  bf16; K = cols 0..127, V = cols 128..255
// out:[B*T, 2048] fp32
// Grid: (T/128, B*NH). Block 256 thr = 8 waves; wave owns 16 query rows.
// K tile staged via global_load_async_to_lds_b128 (ASYNCcnt), V staged
// transposed; staging of block n+1 overlaps compute of block n.
// ---------------------------------------------------------------------------
__global__ __launch_bounds__(256) void attn_fused(
    const __bf16* __restrict__ q, const __bf16* __restrict__ kvb,
    float* __restrict__ out)
{
  __shared__ __align__(64) __bf16 Kt[2][32][HD_];  // K tiles: [buf][s][d]
  __shared__ __align__(64) __bf16 Vt[2][HD_][32];  // V^T tiles: [buf][d][s]
  __shared__ __align__(64) __bf16 Pb[8][16][32];   // per-wave P tile

  const int lane = threadIdx.x & 31;
  const int wv   = threadIdx.x >> 5;
  const int hf   = lane >> 4;
  const int l16  = lane & 15;
  const int qblk = blockIdx.x;
  const int b    = blockIdx.y >> 4;
  const int h    = blockIdx.y & 15;
  const size_t bT = (size_t)b * T_;
  const int qbase = qblk * 128 + wv * 16;

  auto stageK = [&](int buf, int s0) {
#if HAVE_ASYNC_LDS
    #pragma unroll
    for (int r = 0; r < 2; ++r) {
      const int qc = threadIdx.x + r * 256;     // 512 x 16B chunks
      const int s  = qc >> 4;
      const int e  = (qc & 15) * 8;
      const __bf16* g = kvb + (bT + s0 + s) * 256 + e;
      __builtin_amdgcn_global_load_async_to_lds_b128(
          (gas_i32x4*)(uintptr_t)g,
          (las_i32x4*)(uintptr_t)&Kt[buf][s][e],
          0, 0);
    }
#else
    const int s  = threadIdx.x >> 3;
    const int e0 = (threadIdx.x & 7) * 16;
    const __bf16* kr = kvb + (bT + s0 + s) * 256 + e0;
    *(bf16x8*)&Kt[buf][s][e0]     = *(const bf16x8*)(kr);
    *(bf16x8*)&Kt[buf][s][e0 + 8] = *(const bf16x8*)(kr + 8);
#endif
  };
  auto stageV = [&](int buf, int s0) {
    const int s  = threadIdx.x >> 3;
    const int d0 = (threadIdx.x & 7) * 16;
    const __bf16* vr = kvb + (bT + s0 + s) * 256 + 128 + d0;
    bf16x8 v0 = *(const bf16x8*)(vr);
    bf16x8 v1 = *(const bf16x8*)(vr + 8);
    #pragma unroll
    for (int e = 0; e < 8; ++e) { Vt[buf][d0 + e][s] = v0[e]; Vt[buf][d0 + 8 + e][s] = v1[e]; }
  };

  // Q A-fragments: 4 chunks of K=32 covering d = 0..127
  ABf qa[4];
  {
    const __bf16* qr = q + (bT + qbase + l16) * (size_t)C_ + (size_t)h * HD_;
    #pragma unroll
    for (int c = 0; c < 4; ++c) {
      qa[c].v8[0] = *(const bf16x8*)(qr + c * 32 + hf * 16);
      qa[c].v8[1] = *(const bf16x8*)(qr + c * 32 + hf * 16 + 8);
    }
  }

  f32x8 O[8];                       // output accum, 8 d-chunks of 16 cols
  #pragma unroll
  for (int d = 0; d < 8; ++d) O[d] = zero8();
  float mrun[8], lrun[8];           // per-row (vgpr j) softmax state
  #pragma unroll
  for (int j = 0; j < 8; ++j) { mrun[j] = -INFINITY; lrun[j] = 0.f; }

  const int smax = qblk * 128 + 127;  // last key any row in block needs

  // prologue: stage key block 0 into buffer 0
  stageK(0, 0);
  stageV(0, 0);
  wait_async0();
  __syncthreads();

  int ib = 0;
  for (int s0 = 0; s0 <= smax; s0 += 32, ib ^= 1) {
    // issue staging for the next key block into the other buffer first,
    // so it overlaps this block's compute
    const int sn = s0 + 32;
    if (sn <= smax) {
      stageK(ib ^ 1, sn);
      stageV(ib ^ 1, sn);
    }

    // waves whose 16 rows are entirely above this key block skip all math;
    // condition is wave-uniform so EXEC stays all-ones inside (WMMA-safe)
    if (s0 <= qbase + 15) {
      // scores: two 16x16 tiles (keys s0..+15, s0+16..+31) from LDS K tile
      f32x8 st0 = zero8(), st1 = zero8();
      #pragma unroll
      for (int c = 0; c < 4; ++c) {
        const bf16x16 b0 = *(const bf16x16*)(&Kt[ib][l16][c * 32 + hf * 16]);
        const bf16x16 b1 = *(const bf16x16*)(&Kt[ib][16 + l16][c * 32 + hf * 16]);
        st0 = __builtin_amdgcn_wmma_f32_16x16x32_bf16(false, qa[c].v16, false, b0, (short)0, st0, false, false);
        st1 = __builtin_amdgcn_wmma_f32_16x16x32_bf16(false, qa[c].v16, false, b1, (short)0, st1, false, false);
      }

      // causal mask + online softmax; row j in lanes sharing hf, col = l16
      const int k0i = s0 + l16;
      const int k1i = s0 + 16 + l16;
      #pragma unroll
      for (int j = 0; j < 8; ++j) {
        const int qrow = qbase + hf * 8 + j;
        const float s0v = (qrow >= k0i) ? st0[j] : -INFINITY;
        const float s1v = (qrow >= k1i) ? st1[j] : -INFINITY;
        float mx = fmaxf(s0v, s1v);
        #pragma unroll
        for (int sh = 8; sh >= 1; sh >>= 1) mx = fmaxf(mx, __shfl_xor(mx, sh, 32));
        const float mnew = fmaxf(mrun[j], mx);
        const float p0 = __expf(s0v - mnew);
        const float p1 = __expf(s1v - mnew);
        float rs = p0 + p1;
        #pragma unroll
        for (int sh = 8; sh >= 1; sh >>= 1) rs += __shfl_xor(rs, sh, 32);
        const float corr = __expf(mrun[j] - mnew);
        lrun[j] = lrun[j] * corr + rs;
        mrun[j] = mnew;
        #pragma unroll
        for (int d = 0; d < 8; ++d) O[d][j] *= corr;
        const int m = hf * 8 + j;
        Pb[wv][m][l16]      = (__bf16)p0;       // C-layout -> LDS (row-major P)
        Pb[wv][m][16 + l16] = (__bf16)p1;
      }
      __builtin_amdgcn_wave_barrier();          // keep DS stores before DS loads

      // reload P as A-fragment (per-wave DS ops are in-order)
      ABf pa;
      pa.v8[0] = *(const bf16x8*)(&Pb[wv][l16][hf * 16]);
      pa.v8[1] = *(const bf16x8*)(&Pb[wv][l16][hf * 16 + 8]);

      // O[dchunk] += P(16x32) @ V(32x16); B-frag contiguous from transposed Vt
      #pragma unroll
      for (int d = 0; d < 8; ++d) {
        const bf16x16 bv = *(const bf16x16*)(&Vt[ib][d * 16 + l16][hf * 16]);
        O[d] = __builtin_amdgcn_wmma_f32_16x16x32_bf16(false, pa.v16, false, bv, (short)0, O[d], false, false);
      }
    }

    wait_async0();       // own async stores to the next buffer are done
    __syncthreads();     // everyone's staging for the next block is visible
  }

  // normalize and store fp32
  #pragma unroll
  for (int j = 0; j < 8; ++j) {
    const float inv = 1.0f / lrun[j];
    const int row = qbase + hf * 8 + j;
    float* orow = out + (bT + row) * (size_t)C_ + (size_t)h * HD_;
    #pragma unroll
    for (int d = 0; d < 8; ++d) orow[d * 16 + l16] = O[d][j] * inv;
  }
}

// ---------------------------------------------------------------------------
extern "C" void kernel_launch(void* const* d_in, const int* in_sizes, int n_in,
                              void* d_out, int out_size, void* d_ws, size_t ws_size,
                              hipStream_t stream) {
  const float* x    = (const float*)d_in[0];
  const float* w_kv = (const float*)d_in[1];
  const float* w_q  = (const float*)d_in[2];
  float* out = (float*)d_out;

  char* p = (char*)d_ws;
  __bf16* xb   = (__bf16*)p; p += (size_t)B_ * T_ * C_ * 2;        // 16 MB
  __bf16* wqb  = (__bf16*)p; p += (size_t)C_ * C_ * 2;             //  8 MB
  __bf16* wkvb = (__bf16*)p; p += (size_t)2 * HD_ * C_ * 2;        //  1 MB
  __bf16* qb   = (__bf16*)p; p += (size_t)B_ * T_ * C_ * 2;        // 16 MB
  __bf16* kvb  = (__bf16*)p; p += (size_t)B_ * T_ * 2 * HD_ * 2;   //  2 MB

  cvt_f32_bf16<<<4096, 256, 0, stream>>>(x,    xb,   B_ * T_ * C_);
  cvt_f32_bf16<<<4096, 256, 0, stream>>>(w_q,  wqb,  C_ * C_);
  cvt_f32_bf16<<<512,  256, 0, stream>>>(w_kv, wkvb, 2 * HD_ * C_);

  // q = x @ w_q^T, scaled by 1/sqrt(HD) at store
  dim3 gq(C_ / 64, (B_ * T_) / 256);
  gemm_bf16_nt<<<gq, 256, 0, stream>>>(xb, wqb, qb, C_, C_, 0.08838834764831845f);

  // kv = x @ w_kv^T
  dim3 gkv((2 * HD_) / 64, (B_ * T_) / 256);
  gemm_bf16_nt<<<gkv, 256, 0, stream>>>(xb, wkvb, kvb, 2 * HD_, C_, 1.0f);

  // fused causal attention -> fp32 output
  dim3 ga(T_ / 128, B_ * NH_);
  attn_fused<<<ga, 256, 0, stream>>>(qb, kvb, out);
}
